// EfficientUpdateFormer_67216238182570
// MI455X (gfx1250) — matrix operand order repaired and checked by reading, not compiled, not run on hardware
//
#include <hip/hip_runtime.h>
#include <math.h>

// ---------------------------------------------------------------------------
// Problem constants (from setup_inputs)
// ---------------------------------------------------------------------------
#define HD     384
#define KVW    768
#define NH     8
#define DHEAD  48
#define TT     16
#define CIN    320
#define COUTD  130
#define NREAL  3328
#define NSP    256
#define NVV    64
#define NTOT   3392
#define FF     1536
#define LBLK   6
#define DHH    48
#define DWW    64
#define NWH    8            // 48/6
#define NWW    11           // ceil(64/6)  (padded width 66)
#define NWIN   (TT*NWH*NWW) // 1408
#define WSEQ   36
#define NEGF   (-3.402823466e38f)

typedef _Float16 h16;
typedef __attribute__((ext_vector_type(16))) _Float16 v16h;
typedef __attribute__((ext_vector_type(8)))  _Float16 v8h;
typedef __attribute__((ext_vector_type(4)))  _Float16 v4h;
typedef __attribute__((ext_vector_type(8)))  float    v8f;

union U16 { v16h v; _Float16 e[16]; };

__device__ __forceinline__ v8f wmma32(v16h a, v16h b, v8f c) {
  return __builtin_amdgcn_wmma_f32_16x16x32_f16(false, a, false, b, (short)0, c,
                                                false, false);
}

__device__ __forceinline__ float gelu_tanh(float x) {
  const float c0 = 0.7978845608028654f; // sqrt(2/pi)
  float x3 = x * x * x;
  return 0.5f * x * (1.0f + tanhf(c0 * (x + 0.044715f * x3)));
}

__device__ __forceinline__ void cvt8(_Float16* dst, float4 a, float4 b) {
  dst[0] = (_Float16)a.x; dst[1] = (_Float16)a.y;
  dst[2] = (_Float16)a.z; dst[3] = (_Float16)a.w;
  dst[4] = (_Float16)b.x; dst[5] = (_Float16)b.y;
  dst[6] = (_Float16)b.z; dst[7] = (_Float16)b.w;
}

// ---------------------------------------------------------------------------
// GEMM: C[M,N] = act(A[M,K] @ W[K,N] + bias) (+ residual)
// f32 in/out, f16 WMMA inner. 64x64 tile, 256 threads (8 wave32s),
// double-buffered LDS, loads for tile t+1 issued before WMMAs of tile t.
// OOB rows/cols are index-clamped (results discarded in epilogue) so the
// staging loops have no exec-divergent guards.
// ---------------------------------------------------------------------------
__global__ __launch_bounds__(256) void k_gemm(const float* __restrict__ A,
                                              const float* __restrict__ Bw,
                                              const float* __restrict__ bias,
                                              const float* __restrict__ res,
                                              float* __restrict__ C,
                                              int M, int K, int N, int act) {
  __shared__ __align__(16) _Float16 As[2][64 * 40];
  __shared__ __align__(16) _Float16 Bs[2][64 * 40];

  const int tid = threadIdx.x;
  const int wid = tid >> 5;
  const int lane = tid & 31;
  const int hf = lane >> 4;
  const int l16 = lane & 15;
  const int wm = (wid & 3) * 16;   // M sub-offset
  const int wn = (wid >> 2) * 32;  // N sub-offset
  const int m0 = blockIdx.y * 64;
  const int n0 = blockIdx.x * 64;

  // per-thread staging slots
  // A: thread covers 2 quads: e4 = tid + 256*i -> row e4/8, col (e4&7)*4
  // B: thread covers 2 quads: e  = tid + 256*i -> n = e&63, k4 = (e>>6)*4
  float4 ra[2];
  float  rb[8];

  auto loadA = [&](int kk) {
#pragma unroll
    for (int i = 0; i < 2; ++i) {
      int e4 = tid + 256 * i;
      int r = e4 >> 3, c4 = (e4 & 7) * 4;
      int gr = m0 + r; if (gr > M - 1) gr = M - 1;
      ra[i] = *(const float4*)(A + (size_t)gr * K + kk + c4);
    }
  };
  auto loadB = [&](int kk) {
#pragma unroll
    for (int i = 0; i < 2; ++i) {
      int e = tid + 256 * i;
      int n = e & 63, k4 = (e >> 6) * 4;
      int gn = n0 + n; if (gn > N - 1) gn = N - 1;
#pragma unroll
      for (int q = 0; q < 4; ++q)
        rb[i * 4 + q] = Bw[(size_t)(kk + k4 + q) * N + gn];
    }
  };
  auto storeA = [&](int buf) {
#pragma unroll
    for (int i = 0; i < 2; ++i) {
      int e4 = tid + 256 * i;
      int r = e4 >> 3, c4 = (e4 & 7) * 4;
      v4h hv = {(_Float16)ra[i].x, (_Float16)ra[i].y,
                (_Float16)ra[i].z, (_Float16)ra[i].w};
      *(v4h*)&As[buf][r * 40 + c4] = hv;
    }
  };
  auto storeB = [&](int buf) {
#pragma unroll
    for (int i = 0; i < 2; ++i) {
      int e = tid + 256 * i;
      int n = e & 63, k4 = (e >> 6) * 4;
      v4h hv = {(_Float16)rb[i * 4 + 0], (_Float16)rb[i * 4 + 1],
                (_Float16)rb[i * 4 + 2], (_Float16)rb[i * 4 + 3]};
      *(v4h*)&Bs[buf][n * 40 + k4] = hv;
    }
  };

  v8f c0 = {0.f,0.f,0.f,0.f,0.f,0.f,0.f,0.f};
  v8f c1 = c0;

  const int nk = K >> 5;  // all K are multiples of 32
  loadA(0); loadB(0);
  storeA(0); storeB(0);
  __syncthreads();

  for (int t = 0; t < nk; ++t) {
    const int buf = t & 1;
    if (t + 1 < nk) { loadA((t + 1) << 5); loadB((t + 1) << 5); }

    // A fragment: row = wm + l16; K = hf*16 .. hf*16+15
    v16h a;
    {
      const int ar = wm + l16;
      const int akb = hf * 16;
      union { v16h v; v8h h[2]; } u;
      u.h[0] = *(const v8h*)&As[buf][ar * 40 + akb];
      u.h[1] = *(const v8h*)&As[buf][ar * 40 + akb + 8];
      a = u.v;
    }
    // B fragments: col = wn + sub*16 + l16; K = hf*16 .. hf*16+15
    v16h b0, b1;
    {
      const int bkb = hf * 16;
      union { v16h v; v8h h[2]; } u;
      int bc = wn + l16;
      u.h[0] = *(const v8h*)&Bs[buf][bc * 40 + bkb];
      u.h[1] = *(const v8h*)&Bs[buf][bc * 40 + bkb + 8];
      b0 = u.v;
      bc = wn + 16 + l16;
      u.h[0] = *(const v8h*)&Bs[buf][bc * 40 + bkb];
      u.h[1] = *(const v8h*)&Bs[buf][bc * 40 + bkb + 8];
      b1 = u.v;
    }
    c0 = wmma32(a, b0, c0);
    c1 = wmma32(a, b1, c1);

    if (t + 1 < nk) {
      storeA(buf ^ 1);
      storeB(buf ^ 1);
      __syncthreads();
    }
  }

  // epilogue: bias + activation + residual
#pragma unroll
  for (int vv = 0; vv < 8; ++vv) {
    int gm = m0 + wm + vv + (hf << 3);
    if (gm >= M) continue;
    int gn0 = n0 + wn + l16;
    int gn1 = gn0 + 16;
    if (gn0 < N) {
      float x = c0[vv] + (bias ? bias[gn0] : 0.f);
      if (act) x = gelu_tanh(x);
      if (res) x += res[(size_t)gm * N + gn0];
      C[(size_t)gm * N + gn0] = x;
    }
    if (gn1 < N) {
      float x = c1[vv] + (bias ? bias[gn1] : 0.f);
      if (act) x = gelu_tanh(x);
      if (res) x += res[(size_t)gm * N + gn1];
      C[(size_t)gm * N + gn1] = x;
    }
  }
}

// ---------------------------------------------------------------------------
// Flash attention: one wave per (batch, head, 16-query tile).
// Q: [nb,Sq,HD]; KV: [nb,Sk,KVW] (k at head*48, v at 384+head*48); O: [nb,Sq,HD]
// bias = (qok && kok) ? 0 : -FLT_MAX (reference `where(pm,0,NEG)` semantics);
// tile-padding keys are index-clamped and their p forced to 0.
// ---------------------------------------------------------------------------
__global__ __launch_bounds__(32) void k_attn(const float* __restrict__ Q,
                                             const float* __restrict__ KV,
                                             float* __restrict__ O,
                                             const unsigned char* __restrict__ kmask,
                                             const unsigned char* __restrict__ qmask,
                                             int Sq, int Sk, float scale) {
  const int b = blockIdx.z, h = blockIdx.y, qt = blockIdx.x;
  const int lane = threadIdx.x;
  const int hf = lane >> 4;
  const int l16 = lane & 15;
  const int q0 = qt * 16;

  const float* Qb = Q + (size_t)b * Sq * HD + h * DHEAD;
  const float* Kb = KV + (size_t)b * Sk * KVW + h * DHEAD;
  const float* Vb = Kb + HD;

  __shared__ __align__(16) _Float16 Pt[16 * 16];

  // ---- Q fragments (A 16x32 pair; dh=48 zero-padded to 64).
  // Fragment element order == contiguous K runs: frag half = 8 consecutive K.
  U16 qa0, qa1;
  {
    int m = q0 + l16; if (m > Sq - 1) m = Sq - 1;  // clamp: rows independent
    const float* qr = Qb + (size_t)m * HD;
    float4 a0 = *(const float4*)(qr + hf * 8);
    float4 a1 = *(const float4*)(qr + hf * 8 + 4);
    float4 a2 = *(const float4*)(qr + 16 + hf * 8);
    float4 a3 = *(const float4*)(qr + 16 + hf * 8 + 4);
    float4 a4 = *(const float4*)(qr + 32 + hf * 8);   // always < 48
    float4 a5 = *(const float4*)(qr + 32 + hf * 8 + 4);
    cvt8(&qa0.e[0], a0, a1);
    cvt8(&qa0.e[8], a2, a3);
    cvt8(&qa1.e[0], a4, a5);
#pragma unroll
    for (int e = 8; e < 16; ++e) qa1.e[e] = (_Float16)0.f;  // K 48..63 pad
  }

  float rm[8], rl[8];
  bool qok[8];
  int rowi[8];
#pragma unroll
  for (int vv = 0; vv < 8; ++vv) {
    rm[vv] = -INFINITY;
    rl[vv] = 0.f;
    int r = q0 + vv + (hf << 3);
    rowi[vv] = r;
    bool ok = true;
    if (qmask && r < Sq) ok = (qmask[(size_t)b * Sq + r] != 0);
    qok[vv] = ok;
  }

  const v8f zf = {0.f,0.f,0.f,0.f,0.f,0.f,0.f,0.f};
  v8f acc0 = zf, acc1 = zf, acc2 = zf;

  for (int kt0 = 0; kt0 < Sk; kt0 += 16) {
    // ---- K fragments (B 32x16 pair): K halves {hf*8..+7, 16+hf*8..+7}.
    U16 kf0, kf1;
    {
      int kn = kt0 + l16; if (kn > Sk - 1) kn = Sk - 1;  // clamp: p forced 0
      const float* kr = Kb + (size_t)kn * KVW;
      float4 c0a = *(const float4*)(kr + hf * 8);
      float4 c0b = *(const float4*)(kr + hf * 8 + 4);
      float4 c0c = *(const float4*)(kr + 16 + hf * 8);
      float4 c0d = *(const float4*)(kr + 16 + hf * 8 + 4);
      cvt8(&kf0.e[0], c0a, c0b);
      cvt8(&kf0.e[8], c0c, c0d);
      {
        float4 c1a = *(const float4*)(kr + 32 + hf * 8);
        float4 c1b = *(const float4*)(kr + 32 + hf * 8 + 4);
        cvt8(&kf1.e[0], c1a, c1b);
#pragma unroll
        for (int e = 8; e < 16; ++e) kf1.e[e] = (_Float16)0.f;  // K 48..63 pad
      }
    }
    v8f s = wmma32(qa1.v, kf1.v, zf);
    s = wmma32(qa0.v, kf0.v, s);

    const int kcol = kt0 + l16;
    const bool kvalid = (kcol < Sk);
    bool kok = true;
    if (kmask) {
      int ki = kvalid ? kcol : (Sk - 1);
      kok = (kmask[(size_t)b * Sk + ki] != 0);
    }

    float pv[8];
#pragma unroll
    for (int vv = 0; vv < 8; ++vv) {
      float sv = s[vv] * scale;
      if (!(qok[vv] && kok)) sv += NEGF;
      float x = sv;
      x = fmaxf(x, __shfl_xor(x, 1, 32));
      x = fmaxf(x, __shfl_xor(x, 2, 32));
      x = fmaxf(x, __shfl_xor(x, 4, 32));
      x = fmaxf(x, __shfl_xor(x, 8, 32));
      float mnew = fmaxf(rm[vv], x);
      float sc = __expf(rm[vv] - mnew);
      float p = __expf(sv - mnew);
      if (!kvalid) p = 0.f;
      float ps = p;
      ps += __shfl_xor(ps, 1, 32);
      ps += __shfl_xor(ps, 2, 32);
      ps += __shfl_xor(ps, 4, 32);
      ps += __shfl_xor(ps, 8, 32);
      rl[vv] = rl[vv] * sc + ps;
      rm[vv] = mnew;
      acc0[vv] *= sc;
      acc1[vv] *= sc;
      acc2[vv] *= sc;
      pv[vv] = p;
    }

    // transpose P (C-layout) -> A-fragment layout via LDS
    __syncthreads();
#pragma unroll
    for (int vv = 0; vv < 8; ++vv)
      Pt[(vv + (hf << 3)) * 16 + l16] = (_Float16)pv[vv];
    __syncthreads();

    U16 pa;
#pragma unroll
    for (int vv = 0; vv < 8; ++vv) {
      const int kb = (vv << 1) + (hf << 4);
#pragma unroll
      for (int pp = 0; pp < 2; ++pp) {
        int k = kb + pp;
        pa.e[2 * vv + pp] = (k < 16) ? Pt[l16 * 16 + k] : (_Float16)0.f;
      }
    }

    // ---- V fragments (dh=48 -> three 16-wide N tiles).
    // Only lanes with hf==0 carry real K (0..15); hf==1 pairs with zero P.
#pragma unroll
    for (int j = 0; j < 3; ++j) {
      U16 vf;
      const int d = j * 16 + l16;
      if (hf == 0) {
        float tv[16];
#pragma unroll
        for (int e = 0; e < 16; ++e) {
          int gk = kt0 + e; if (gk > Sk - 1) gk = Sk - 1;  // p==0 there
          tv[e] = Vb[(size_t)gk * KVW + d];
        }
#pragma unroll
        for (int e = 0; e < 16; ++e) vf.e[e] = (_Float16)tv[e];
      } else {
#pragma unroll
        for (int e = 0; e < 16; ++e) vf.e[e] = (_Float16)0.f;
      }
      if (j == 0)      acc0 = wmma32(pa.v, vf.v, acc0);
      else if (j == 1) acc1 = wmma32(pa.v, vf.v, acc1);
      else             acc2 = wmma32(pa.v, vf.v, acc2);
    }
  }

#pragma unroll
  for (int vv = 0; vv < 8; ++vv) {
    int r = rowi[vv];
    if (r < Sq) {
      float inv = 1.f / rl[vv];
      size_t base = ((size_t)b * Sq + r) * HD + h * DHEAD + l16;
      O[base]      = acc0[vv] * inv;
      O[base + 16] = acc1[vv] * inv;
      O[base + 32] = acc2[vv] * inv;
    }
  }
}

// ---------------------------------------------------------------------------
// LayerNorm over last dim (384), optional affine, one row per 128-thr block.
// ---------------------------------------------------------------------------
__global__ __launch_bounds__(128) void k_ln(const float* __restrict__ X,
                                            float* __restrict__ Y,
                                            const float* __restrict__ g,
                                            const float* __restrict__ bb,
                                            float eps) {
  const int row = blockIdx.x;
  const int t = threadIdx.x;
  const float* x = X + (size_t)row * HD;
  float* y = Y + (size_t)row * HD;
  __shared__ float red[128];
  float v0 = x[t], v1 = x[t + 128], v2 = x[t + 256];
  red[t] = v0 + v1 + v2;
  __syncthreads();
  for (int o = 64; o > 0; o >>= 1) {
    if (t < o) red[t] += red[t + o];
    __syncthreads();
  }
  float mean = red[0] * (1.f / 384.f);
  __syncthreads();
  float d0 = v0 - mean, d1 = v1 - mean, d2 = v2 - mean;
  red[t] = d0 * d0 + d1 * d1 + d2 * d2;
  __syncthreads();
  for (int o = 64; o > 0; o >>= 1) {
    if (t < o) red[t] += red[t + o];
    __syncthreads();
  }
  float rstd = rsqrtf(red[0] * (1.f / 384.f) + eps);
  float g0 = g ? g[t] : 1.f, g1 = g ? g[t + 128] : 1.f, g2 = g ? g[t + 256] : 1.f;
  float b0 = bb ? bb[t] : 0.f, b1 = bb ? bb[t + 128] : 0.f, b2 = bb ? bb[t + 256] : 0.f;
  y[t]       = d0 * rstd * g0 + b0;
  y[t + 128] = d1 * rstd * g1 + b1;
  y[t + 256] = d2 * rstd * g2 + b2;
}

// ---------------------------------------------------------------------------
// Row-copy helpers (rows of 384 f32 = 96 float4 per 96-thread block)
// ---------------------------------------------------------------------------
__global__ __launch_bounds__(96) void k_copy_rows(const float* __restrict__ src,
                                                  float* __restrict__ dst,
                                                  int rows_per_group,
                                                  long long sgs, long long dgs) {
  int r = blockIdx.x;
  int g = r / rows_per_group, rr = r % rows_per_group;
  const float4* s = (const float4*)(src + (size_t)g * sgs + (size_t)rr * HD);
  float4* d = (float4*)(dst + (size_t)g * dgs + (size_t)rr * HD);
  d[threadIdx.x] = s[threadIdx.x];
}

// dst[(b*A + a)*HD] = src[(a*Bi + b)*HD]
__global__ __launch_bounds__(96) void k_transpose(const float* __restrict__ src,
                                                  float* __restrict__ dst,
                                                  int A, int Bi) {
  int r = blockIdx.x;
  int a = r / Bi, b = r % Bi;
  const float4* s = (const float4*)(src + ((size_t)a * Bi + b) * HD);
  float4* d = (float4*)(dst + ((size_t)b * A + a) * HD);
  d[threadIdx.x] = s[threadIdx.x];
}

// dense (t, 48x64) -> windows (1408, 36) with zero pad (width padded to 66)
__global__ __launch_bounds__(96) void k_win_gather(const float* __restrict__ R,
                                                   float* __restrict__ W) {
  int row = blockIdx.x;
  int w = row / WSEQ, i = row % WSEQ;
  int t = w / (NWH * NWW), rem = w % (NWH * NWW);
  int wh = rem / NWW, ww = rem % NWW;
  int hh = wh * 6 + i / 6;
  int cc = ww * 6 + i % 6;
  float4* d = (float4*)(W + (size_t)row * HD);
  if (hh < DHH && cc < DWW) {
    const float4* s =
        (const float4*)(R + ((size_t)t * NREAL + NSP + (size_t)hh * DWW + cc) * HD);
    d[threadIdx.x] = s[threadIdx.x];
  } else {
    d[threadIdx.x] = make_float4(0.f, 0.f, 0.f, 0.f);
  }
}

__global__ __launch_bounds__(96) void k_win_scatter(const float* __restrict__ W,
                                                    float* __restrict__ R) {
  int row = blockIdx.x;
  int w = row / WSEQ, i = row % WSEQ;
  int t = w / (NWH * NWW), rem = w % (NWH * NWW);
  int wh = rem / NWW, ww = rem % NWW;
  int hh = wh * 6 + i / 6;
  int cc = ww * 6 + i % 6;
  if (hh < DHH && cc < DWW) {
    const float4* s = (const float4*)(W + (size_t)row * HD);
    float4* d =
        (float4*)(R + ((size_t)t * NREAL + NSP + (size_t)hh * DWW + cc) * HD);
    d[threadIdx.x] = s[threadIdx.x];
  }
}

// mask[row] = (sum |x| > 0)
__global__ __launch_bounds__(128) void k_rowmask(const float* __restrict__ X,
                                                 unsigned char* __restrict__ M) {
  int row = blockIdx.x;
  const float* x = X + (size_t)row * HD;
  int t = threadIdx.x;
  __shared__ float red[128];
  red[t] = fabsf(x[t]) + fabsf(x[t + 128]) + fabsf(x[t + 256]);
  __syncthreads();
  for (int o = 64; o > 0; o >>= 1) {
    if (t < o) red[t] += red[t + o];
    __syncthreads();
  }
  if (t == 0) M[row] = (red[0] > 0.f) ? 1 : 0;
}

// ---------------------------------------------------------------------------
// Host-side orchestration helpers
// ---------------------------------------------------------------------------
struct SAP {
  const float *qw, *qb, *kvw, *kvb, *ow, *ob, *f1w, *f1b, *f2w, *f2b;
};
struct CAP {
  SAP s;
  const float *ncw, *ncb;
};
struct WSB {
  float *H, *Q, *KV, *O, *MID;
};

static void run_gemm(const float* A, const float* W, const float* bias, float* C,
                     int M, int K, int N, int act, const float* res,
                     hipStream_t st) {
  dim3 g((N + 63) / 64, (M + 63) / 64, 1);
  k_gemm<<<g, 256, 0, st>>>(A, W, bias, res, C, M, K, N, act);
}

static void run_attn(const float* Q, const float* KV, float* O,
                     const unsigned char* km, const unsigned char* qm, int nb,
                     int Sq, int Sk, hipStream_t st) {
  dim3 g((Sq + 15) / 16, NH, nb);
  k_attn<<<g, 32, 0, st>>>(Q, KV, O, km, qm, Sq, Sk, 0.14433756729740643f);
}

static void run_mlp(const float* Hin, const SAP& p, float* X, int rows,
                    const WSB& w, hipStream_t st) {
  const int CH = 8192;
  for (int r0 = 0; r0 < rows; r0 += CH) {
    int cr = (rows - r0 < CH) ? (rows - r0) : CH;
    run_gemm(Hin + (size_t)r0 * HD, p.f1w, p.f1b, w.MID, cr, HD, FF, 1, nullptr, st);
    run_gemm(w.MID, p.f2w, p.f2b, X + (size_t)r0 * HD, cr, FF, HD, 0,
             X + (size_t)r0 * HD, st);
  }
}

static void run_self(float* X, int nb, int S, const SAP& p,
                     const unsigned char* msk, const WSB& w, hipStream_t st) {
  int rows = nb * S;
  k_ln<<<rows, 128, 0, st>>>(X, w.H, nullptr, nullptr, 1e-6f);
  run_gemm(w.H, p.qw, p.qb, w.Q, rows, HD, HD, 0, nullptr, st);
  run_gemm(w.H, p.kvw, p.kvb, w.KV, rows, HD, KVW, 0, nullptr, st);
  run_attn(w.Q, w.KV, w.O, msk, msk, nb, S, S, st);
  run_gemm(w.O, p.ow, p.ob, X, rows, HD, HD, 0, X, st);
  k_ln<<<rows, 128, 0, st>>>(X, w.H, nullptr, nullptr, 1e-6f);
  run_mlp(w.H, p, X, rows, w, st);
}

static void run_cross(float* X, int nb, int Sq, const float* CTX, int Sk,
                      const CAP& c, const unsigned char* km,
                      const unsigned char* qm, const WSB& w, hipStream_t st) {
  int rq = nb * Sq, rk = nb * Sk;
  k_ln<<<rk, 128, 0, st>>>(CTX, w.H, c.ncw, c.ncb, 1e-5f);
  run_gemm(w.H, c.s.kvw, c.s.kvb, w.KV, rk, HD, KVW, 0, nullptr, st);
  k_ln<<<rq, 128, 0, st>>>(X, w.H, nullptr, nullptr, 1e-6f);
  run_gemm(w.H, c.s.qw, c.s.qb, w.Q, rq, HD, HD, 0, nullptr, st);
  run_attn(w.Q, w.KV, w.O, km, qm, nb, Sq, Sk, st);
  run_gemm(w.O, c.s.ow, c.s.ob, X, rq, HD, HD, 0, X, st);
  k_ln<<<rq, 128, 0, st>>>(X, w.H, nullptr, nullptr, 1e-6f);
  run_mlp(w.H, c.s, X, rq, w, st);
}

// ---------------------------------------------------------------------------
extern "C" void kernel_launch(void* const* d_in, const int* in_sizes, int n_in,
                              void* d_out, int out_size, void* d_ws,
                              size_t ws_size, hipStream_t stream) {
  (void)in_sizes; (void)n_in; (void)out_size;

  const float* input = (const float*)d_in[0];
  const unsigned char* amask = (const unsigned char*)d_in[1];
  const float* w_in = (const float*)d_in[2];
  const float* b_in = (const float*)d_in[3];
  const float* w_out = (const float*)d_in[4];
  const float* b_out = (const float*)d_in[5];
  const float* vtracks = (const float*)d_in[6];
  const float* sa_qw = (const float*)d_in[7];
  const float* sa_qb = (const float*)d_in[8];
  const float* sa_kvw = (const float*)d_in[9];
  const float* sa_kvb = (const float*)d_in[10];
  const float* sa_ow = (const float*)d_in[11];
  const float* sa_ob = (const float*)d_in[12];
  const float* sa_f1w = (const float*)d_in[13];
  const float* sa_f1b = (const float*)d_in[14];
  const float* sa_f2w = (const float*)d_in[15];
  const float* sa_f2b = (const float*)d_in[16];
  const float* ca_qw = (const float*)d_in[17];
  const float* ca_qb = (const float*)d_in[18];
  const float* ca_kvw = (const float*)d_in[19];
  const float* ca_kvb = (const float*)d_in[20];
  const float* ca_ow = (const float*)d_in[21];
  const float* ca_ob = (const float*)d_in[22];
  const float* ca_f1w = (const float*)d_in[23];
  const float* ca_f1b = (const float*)d_in[24];
  const float* ca_f2w = (const float*)d_in[25];
  const float* ca_f2b = (const float*)d_in[26];
  const float* ca_ncw = (const float*)d_in[27];
  const float* ca_ncb = (const float*)d_in[28];

  // ---- workspace carve ----
  float* base = (float*)d_ws;
  size_t off = 0;
  auto take = [&](size_t n) { float* p = base + off; off += n; return p; };
  const size_t RT = (size_t)NTOT * TT;   // 54272 rows
  const size_t RR = (size_t)NREAL * TT;  // 53248 rows

  float* Xt = take(RT * HD);
  float* Xs = take(RT * HD);
  float* R  = take(RR * HD);
  float* V  = take((size_t)NVV * TT * HD);
  WSB w;
  w.H   = take(RT * HD);
  w.Q   = take(RT * HD);
  w.KV  = take(RT * KVW);
  w.O   = take(RT * HD);
  w.MID = take((size_t)8192 * FF);
  unsigned char* Mb = (unsigned char*)(base + off);
  size_t need = off * sizeof(float) + (size_t)NWIN * WSEQ;
  if (ws_size < need) return;  // deterministic no-op if scratch too small
  float* Wb = Xt;  // windows buffer aliases the (idle) time-major tokens

  auto sap = [&](int s, int j) {
    SAP p; size_t L = (size_t)(s * LBLK + j);
    p.qw = sa_qw + L * HD * HD;   p.qb = sa_qb + L * HD;
    p.kvw = sa_kvw + L * HD * KVW; p.kvb = sa_kvb + L * KVW;
    p.ow = sa_ow + L * HD * HD;   p.ob = sa_ob + L * HD;
    p.f1w = sa_f1w + L * HD * FF; p.f1b = sa_f1b + L * FF;
    p.f2w = sa_f2w + L * FF * HD; p.f2b = sa_f2b + L * HD;
    return p;
  };
  auto cap = [&](int s, int j) {
    CAP c; size_t L = (size_t)(s * LBLK + j);
    c.s.qw = ca_qw + L * HD * HD;   c.s.qb = ca_qb + L * HD;
    c.s.kvw = ca_kvw + L * HD * KVW; c.s.kvb = ca_kvb + L * KVW;
    c.s.ow = ca_ow + L * HD * HD;   c.s.ob = ca_ob + L * HD;
    c.s.f1w = ca_f1w + L * HD * FF; c.s.f1b = ca_f1b + L * FF;
    c.s.f2w = ca_f2w + L * FF * HD; c.s.f2b = ca_f2b + L * HD;
    c.ncw = ca_ncw + L * HD;        c.ncb = ca_ncb + L * HD;
    return c;
  };

  // ---- input projection: real = input @ w_in + b_in ----
  run_gemm(input, w_in, b_in, R, TT * NREAL, CIN, HD, 0, nullptr, stream);
  // assemble space-major tokens Xs = [real | virt] per t, then time-major Xt
  k_copy_rows<<<TT * NREAL, 96, 0, stream>>>(R, Xs, NREAL,
      (long long)NREAL * HD, (long long)NTOT * HD);
  k_copy_rows<<<TT * NVV, 96, 0, stream>>>(vtracks, Xs + (size_t)NREAL * HD, NVV,
      0LL, (long long)NTOT * HD);
  k_transpose<<<(int)RT, 96, 0, stream>>>(Xs, Xt, TT, NTOT);

  for (int j = 0; j < LBLK; ++j) {
    SAP sa0 = sap(0, j), sa1 = sap(1, j), sa2 = sap(2, j);
    CAP ca0 = cap(0, j), ca1 = cap(1, j);

    // 1) temporal self-attention over all tracks
    run_self(Xt, NTOT, TT, sa0, nullptr, w, stream);

    // 2) to space-major; split real / virtual
    k_transpose<<<(int)RT, 96, 0, stream>>>(Xt, Xs, NTOT, TT);
    k_copy_rows<<<TT * NREAL, 96, 0, stream>>>(Xs, R, NREAL,
        (long long)NTOT * HD, (long long)NREAL * HD);
    k_copy_rows<<<TT * NVV, 96, 0, stream>>>(Xs + (size_t)NREAL * HD, V, NVV,
        (long long)NTOT * HD, (long long)NVV * HD);

    // 3) virt <- cross(real), key mask
    run_cross(V, TT, NVV, R, NREAL, ca0, amask, nullptr, w, stream);
    // 4) virt self-attention
    run_self(V, TT, NVV, sa1, nullptr, w, stream);

    // 5) local windowed attention on dense tokens
    k_win_gather<<<NWIN * WSEQ, 96, 0, stream>>>(R, Wb);
    k_rowmask<<<NWIN * WSEQ, 128, 0, stream>>>(Wb, Mb);
    run_self(Wb, NWIN, WSEQ, sa2, Mb, w, stream);
    k_win_scatter<<<NWIN * WSEQ, 96, 0, stream>>>(Wb, R);

    // 6) real <- cross(virt), query mask
    run_cross(R, TT, NREAL, V, NVV, ca1, nullptr, amask, w, stream);

    // 7) reassemble and go back to time-major (except last block)
    if (j != LBLK - 1) {
      k_copy_rows<<<TT * NREAL, 96, 0, stream>>>(R, Xs, NREAL,
          (long long)NREAL * HD, (long long)NTOT * HD);
      k_copy_rows<<<TT * NVV, 96, 0, stream>>>(V, Xs + (size_t)NREAL * HD, NVV,
          (long long)NVV * HD, (long long)NTOT * HD);
      k_transpose<<<(int)RT, 96, 0, stream>>>(Xs, Xt, TT, NTOT);
    }
  }

  // ---- output projection ----
  run_gemm(R, w_out, b_out, (float*)d_out, TT * NREAL, HD, COUTD, 0, nullptr,
           stream);
}
